// STColAttention_4303557231001
// MI455X (gfx1250) — matrix operand order repaired and checked by reading, not compiled, not run on hardware
//
#include <hip/hip_runtime.h>
#include <math.h>

// ---- problem constants (from reference) ----
#define Bn  8
#define Sn  100
#define Tn  256
#define Cn  64
#define Hn  256      // HID
#define An  128      // AHS = 4 heads * 32
#define Tt  32       // T-tile processed per iteration

typedef __bf16 bf16;
typedef __attribute__((ext_vector_type(16))) __bf16 v16bf;
typedef __attribute__((ext_vector_type(4)))  __bf16 v4bf;
typedef __attribute__((ext_vector_type(8)))  float  v8f;
typedef __attribute__((ext_vector_type(4)))  float  v4f;

// ---------------- helpers ----------------
// native convert: lowers to hardware bf16 cvt instead of bit-twiddled RNE
__device__ inline bf16 f2bf(float x) { return (bf16)x; }

__device__ inline float softplusf(float x) {
  return (x > 20.f) ? x : log1pf(__expf(x));
}

// A fragment: 16x32 bf16, row-major source with leading dim lda.
// ISA 7.12.2: lane m=lane&15; lanes<16 hold k={0..7,16..23}, lanes>=16 k={8..15,24..31}
__device__ inline v16bf load_frag_a(const bf16* a, int lda, int lane) {
  const int m  = lane & 15;
  const int kh = (lane >> 4) << 3;          // 0 or 8
  const bf16* row = a + m * lda + kh;
  v16bf f;
#pragma unroll
  for (int e = 0; e < 8; ++e) f[e] = row[e];
#pragma unroll
  for (int e = 0; e < 8; ++e) f[8 + e] = row[16 + e];
  return f;
}

// B fragment: 32x16 bf16 where B[k][n] = src[n*ldb + k] (i.e. per-lane row is
// contiguous in k).  Lanes 0-15: n, k={0..7,16..23}; lanes 16-31: n, k={8..15,24..31}.
__device__ inline v16bf load_frag_b(const bf16* bsrc, int ldb, int lane) {
  const int n  = lane & 15;
  const int k0 = (lane >> 4) << 3;          // 0 or 8
  const bf16* row = bsrc + n * ldb + k0;
  v16bf f;
#pragma unroll
  for (int e = 0; e < 8; ++e) f[e] = row[e];
#pragma unroll
  for (int e = 0; e < 8; ++e) f[8 + e] = row[16 + e];
  return f;
}

#define WMMA_BF16(a, b, c) \
  __builtin_amdgcn_wmma_f32_16x16x32_bf16(false, (a), false, (b), (short)0, (c), false, false)

// ---------------- kernel 1: FIRE bias tables + bf16 weight conversion ----------------
__device__ inline float fire_eval(float d, float logc, const float* __restrict__ w1,
                                  const float* __restrict__ w2, float maxd) {
  float c = softplusf(logc) + 1e-4f;
  float x = log1pf(c * d) / log1pf(c * maxd);
  float acc = 0.f;
#pragma unroll
  for (int j = 0; j < 32; ++j) {
    float z = x * w1[j];                     // w1 is (32,1)
    acc += (z / (1.f + __expf(-z))) * w2[j]; // silu, w2 is (1,32)
  }
  return acc;
}

__global__ void fire_prep_kernel(
    const float* __restrict__ st,
    const float* __restrict__ Wq, const float* __restrict__ Wk,
    const float* __restrict__ Wv, const float* __restrict__ Wo,
    const float* __restrict__ s_logc, const float* __restrict__ s_w1, const float* __restrict__ s_w2,
    const float* __restrict__ t_logc, const float* __restrict__ t_w1, const float* __restrict__ t_w2,
    float* __restrict__ sp, float* __restrict__ tp,
    bf16* __restrict__ wq_bf, bf16* __restrict__ wk_bf,
    bf16* __restrict__ wv_bf, bf16* __restrict__ wo_bf) {
  const int i = blockIdx.x * blockDim.x + threadIdx.x;
  if (i < An * Hn / 4) {                     // all four weight mats are 32768 elems
    const v4f q = *(const v4f*)(Wq + 4 * i);
    const v4f k = *(const v4f*)(Wk + 4 * i);
    const v4f v = *(const v4f*)(Wv + 4 * i);
    const v4f o = *(const v4f*)(Wo + 4 * i);
    v4bf qb, kb, vb, ob;
#pragma unroll
    for (int e = 0; e < 4; ++e) {
      qb[e] = f2bf(q[e]); kb[e] = f2bf(k[e]); vb[e] = f2bf(v[e]); ob[e] = f2bf(o[e]);
    }
    *(v4bf*)(wq_bf + 4 * i) = qb;
    *(v4bf*)(wk_bf + 4 * i) = kb;
    *(v4bf*)(wv_bf + 4 * i) = vb;
    *(v4bf*)(wo_bf + 4 * i) = ob;
  }
  if (i < Bn * Tn * Cn) {
    sp[i] = fire_eval(st[2 * i + 0], s_logc[0], s_w1, s_w2, 180.f);
    tp[i] = fire_eval(st[2 * i + 1], t_logc[0], t_w1, t_w2, 365.f);
  }
}

// stage 32x256 f32 tile from global into LDS as bf16, 128-bit loads
__device__ inline void stage_tile_bf16(bf16* __restrict__ dst,
                                       const float* __restrict__ src, int tid) {
  for (int i = tid * 4; i < 32 * Hn; i += 256 * 4) {
    const v4f x = *(const v4f*)(src + i);
    v4bf y;
#pragma unroll
    for (int e = 0; e < 4; ++e) y[e] = f2bf(x[e]);
    *(v4bf*)(dst + i) = y;
  }
}

// ---------------- kernel 2: fused attention, one block per (b,s) ----------------
__global__ void __launch_bounds__(256)
st_attn_kernel(const float* __restrict__ hidden,   // (B,S,T,H)
               const float* __restrict__ srcemb,   // (B,S,C,H)
               const float* __restrict__ mask,     // (B,1,1,T,C)
               const float* __restrict__ sp,       // (B,T,C)
               const float* __restrict__ tp,       // (B,T,C)
               const bf16*  __restrict__ wq, const bf16* __restrict__ wk,
               const bf16*  __restrict__ wv, const bf16* __restrict__ wo,
               const float* __restrict__ bq, const float* __restrict__ bk,
               const float* __restrict__ bv, const float* __restrict__ bo,
               const float* __restrict__ s_scale, const float* __restrict__ t_scale,
               float* __restrict__ out) {
  // 64KB static LDS, heavily overlaid:
  //   [0,16K)   Xh: 32x256 bf16 staging   (overlaid later by P: 32x64 bf16)
  //   [16K,32K) K : 64x128 bf16
  //   [32K,48K) Vt: 128x64 bf16 (V transposed, d-major)
  //   [48K,56K) Qc: 32x128 bf16 (Q, overwritten per-head by ctx)
  //   [56K,64K) S : 32x64 f32 per-head scores
  __shared__ __align__(16) unsigned char smem[65536];
  bf16*  Xh = (bf16*)smem;
  bf16*  Pl = (bf16*)smem;
  bf16*  Kl = (bf16*)(smem + 16384);
  bf16*  Vt = (bf16*)(smem + 32768);
  bf16*  Qc = (bf16*)(smem + 49152);
  float* Sl = (float*)(smem + 57344);

  const int tid  = threadIdx.x;
  const int wave = tid >> 5;
  const int lane = tid & 31;
  const int b = blockIdx.x / Sn;
  const int s = blockIdx.x % Sn;

  const float ss = softplusf(s_scale[s]) + 1e-4f;
  const float ts = softplusf(t_scale[s]) + 1e-4f;

  const int n_out = lane & 15;
  const int mh    = (lane >> 4) << 3;       // D-frag: m = r + mh

  // ---- Phase A: K = Xs@Wk^T + bk (row-major), V = Xs@Wv^T + bv (transposed) ----
  const float* xs = srcemb + ((size_t)(b * Sn + s)) * Cn * Hn;
  for (int ch = 0; ch < 2; ++ch) {          // two halves of the 64 context rows
    __syncthreads();
    stage_tile_bf16(Xh, xs + ch * 32 * Hn, tid);
    __syncthreads();
    for (int tile = wave; tile < 32; tile += 8) {
      const int which = tile >> 4;          // 0 = K, 1 = V
      const int mt = (tile >> 3) & 1;
      const int nt = tile & 7;
      const bf16* W = which ? wv : wk;
      v8f acc = {};
#pragma unroll
      for (int kk = 0; kk < Hn; kk += 32) {
        v16bf a  = load_frag_a(Xh + mt * 16 * Hn + kk, Hn, lane);
        v16bf bm = load_frag_b(W + nt * 16 * Hn + kk, Hn, lane);
        acc = WMMA_BF16(a, bm, acc);
      }
      const int ng    = nt * 16 + n_out;
      const int crow0 = ch * 32 + mt * 16 + mh;
      if (which == 0) {
        const float bias = bk[ng];
#pragma unroll
        for (int r = 0; r < 8; ++r) Kl[(crow0 + r) * An + ng] = f2bf(acc[r] + bias);
      } else {
        const float bias = bv[ng];
#pragma unroll
        for (int r = 0; r < 8; ++r) Vt[ng * Cn + crow0 + r] = f2bf(acc[r] + bias);
      }
    }
  }

  // ---- Phase B: loop over T tiles ----
  const float* xhg  = hidden + ((size_t)(b * Sn + s)) * Tn * Hn;
  float*       outg = out    + ((size_t)(b * Sn + s)) * Tn * Hn;

  for (int t0 = 0; t0 < Tn; t0 += Tt) {
    __syncthreads();
    stage_tile_bf16(Xh, xhg + (size_t)t0 * Hn, tid);
    // prefetch next T-tile while we crunch this one (global_prefetch_b8)
    if (t0 + Tt < Tn)
      __builtin_prefetch(xhg + (size_t)(t0 + Tt) * Hn + tid * 32, 0, 3);
    __syncthreads();

    // Q projection: (32x256)@(256x128), 16 tiles / 8 waves
    for (int tile = wave; tile < 16; tile += 8) {
      const int mt = tile >> 3, nt = tile & 7;
      v8f acc = {};
#pragma unroll
      for (int kk = 0; kk < Hn; kk += 32) {
        v16bf a  = load_frag_a(Xh + mt * 16 * Hn + kk, Hn, lane);
        v16bf bm = load_frag_b(wq + nt * 16 * Hn + kk, Hn, lane);
        acc = WMMA_BF16(a, bm, acc);
      }
      const int ng = nt * 16 + n_out;
      const float bias = bq[ng];
#pragma unroll
      for (int r = 0; r < 8; ++r) Qc[(mt * 16 + mh + r) * An + ng] = f2bf(acc[r] + bias);
    }
    __syncthreads();

    // per-head: scores -> bias+softmax -> ctx (overwrites Q's columns for head h)
    for (int h = 0; h < 4; ++h) {
      // scores: Qh (32x32) @ Kh^T (32x64): 8 tiles / 8 waves, one K-step each
      {
        const int mt = wave >> 2, nt = wave & 3;
        v8f acc = {};
        v16bf a  = load_frag_a(Qc + mt * 16 * An + 32 * h, An, lane);
        v16bf bm = load_frag_b(Kl + nt * 16 * An + 32 * h, An, lane);
        acc = WMMA_BF16(a, bm, acc);
#pragma unroll
        for (int r = 0; r < 8; ++r)
          Sl[(mt * 16 + mh + r) * Cn + nt * 16 + n_out] = acc[r] * 0.17677669529663687f;
      }
      __syncthreads();
      // add FIRE bias + attention mask
      for (int i = tid; i < Tt * Cn; i += 256) {
        const int t = i >> 6, c = i & 63;
        const size_t gi = ((size_t)b * Tn + (t0 + t)) * Cn + c;
        Sl[i] += sp[gi] * ss + tp[gi] * ts + mask[gi];
      }
      __syncthreads();
      // softmax over C=64; 4 lanes per row, reduce with wave32 shuffles
      if (tid < Tt * 4) {
        const int row = tid >> 2, q = tid & 3;
        const float* srow = Sl + row * Cn + q * 16;
        float mx = -3.0e38f;
#pragma unroll
        for (int e = 0; e < 16; ++e) mx = fmaxf(mx, srow[e]);
        mx = fmaxf(mx, __shfl_xor(mx, 1));
        mx = fmaxf(mx, __shfl_xor(mx, 2));
        float ev[16], sum = 0.f;
#pragma unroll
        for (int e = 0; e < 16; ++e) { ev[e] = __expf(srow[e] - mx); sum += ev[e]; }
        sum += __shfl_xor(sum, 1);
        sum += __shfl_xor(sum, 2);
        const float rs = 1.0f / sum;
#pragma unroll
        for (int e = 0; e < 16; ++e) Pl[row * Cn + q * 16 + e] = f2bf(ev[e] * rs);
      }
      __syncthreads();
      // ctx: P (32x64) @ Vh (64x32) -> Qc columns [32h, 32h+32)
      if (wave < 4) {
        const int mt = wave >> 1, nt = wave & 1;
        v8f acc = {};
#pragma unroll
        for (int kk = 0; kk < Cn; kk += 32) {
          v16bf a  = load_frag_a(Pl + mt * 16 * Cn + kk, Cn, lane);
          v16bf bm = load_frag_b(Vt + (32 * h + nt * 16) * Cn + kk, Cn, lane);
          acc = WMMA_BF16(a, bm, acc);
        }
#pragma unroll
        for (int r = 0; r < 8; ++r)
          Qc[(mt * 16 + mh + r) * An + 32 * h + nt * 16 + n_out] = f2bf(acc[r]);
      }
      __syncthreads();
    }

    // output projection: ctx (32x128) @ Wo^T (128x256) + bo -> global
    for (int tile = wave; tile < 32; tile += 8) {
      const int mt = tile >> 4, nt = tile & 15;
      v8f acc = {};
#pragma unroll
      for (int kk = 0; kk < An; kk += 32) {
        v16bf a  = load_frag_a(Qc + mt * 16 * An + kk, An, lane);
        v16bf bm = load_frag_b(wo + nt * 16 * An + kk, An, lane);
        acc = WMMA_BF16(a, bm, acc);
      }
      const int ng = nt * 16 + n_out;
      const float bias = bo[ng];
#pragma unroll
      for (int r = 0; r < 8; ++r)
        outg[(size_t)(t0 + mt * 16 + mh + r) * Hn + ng] = acc[r] + bias;
    }
  }
}

// ---------------- launch ----------------
extern "C" void kernel_launch(void* const* d_in, const int* in_sizes, int n_in,
                              void* d_out, int out_size, void* d_ws, size_t ws_size,
                              hipStream_t stream) {
  (void)in_sizes; (void)n_in; (void)out_size; (void)ws_size;
  const float* hidden  = (const float*)d_in[0];
  const float* srcemb  = (const float*)d_in[1];
  const float* mask    = (const float*)d_in[2];
  const float* st      = (const float*)d_in[3];
  const float* Wq      = (const float*)d_in[4];
  const float* bq      = (const float*)d_in[5];
  const float* Wk      = (const float*)d_in[6];
  const float* bk      = (const float*)d_in[7];
  const float* Wv      = (const float*)d_in[8];
  const float* bv      = (const float*)d_in[9];
  const float* Wo      = (const float*)d_in[10];
  const float* bo      = (const float*)d_in[11];
  const float* s_logc  = (const float*)d_in[12];
  const float* s_w1    = (const float*)d_in[13];
  const float* s_w2    = (const float*)d_in[14];
  const float* t_logc  = (const float*)d_in[15];
  const float* t_w1    = (const float*)d_in[16];
  const float* t_w2    = (const float*)d_in[17];
  const float* s_scale = (const float*)d_in[18];
  const float* t_scale = (const float*)d_in[19];

  // workspace layout (~1.25 MB total)
  unsigned char* ws = (unsigned char*)d_ws;
  bf16*  wq_bf = (bf16*)(ws + 0);
  bf16*  wk_bf = (bf16*)(ws + 65536);
  bf16*  wv_bf = (bf16*)(ws + 131072);
  bf16*  wo_bf = (bf16*)(ws + 196608);
  float* sp    = (float*)(ws + 262144);
  float* tp    = (float*)(ws + 262144 + 524288);

  const int prepN = Bn * Tn * Cn;            // 131072 FIRE evaluations
  fire_prep_kernel<<<(prepN + 255) / 256, 256, 0, stream>>>(
      st, Wq, Wk, Wv, Wo, s_logc, s_w1, s_w2, t_logc, t_w1, t_w2,
      sp, tp, wq_bf, wk_bf, wv_bf, wo_bf);

  st_attn_kernel<<<Bn * Sn, 256, 0, stream>>>(
      hidden, srcemb, mask, sp, tp,
      wq_bf, wk_bf, wv_bf, wo_bf,
      bq, bk, bv, bo, s_scale, t_scale,
      (float*)d_out);
}
